// RNN_15676630630599
// MI455X (gfx1250) — matrix operand order repaired and checked by reading, not compiled, not run on hardware
//
#include <hip/hip_runtime.h>
#include <hip/hip_bf16.h>

// ---------------------------------------------------------------------------
// Problem constants (from reference): T=65536, H=256, C=10, I=6
// ---------------------------------------------------------------------------
#define TT 65536
#define HH 256
#define CC 10
#define II 6

typedef __attribute__((ext_vector_type(16))) _Float16 v16h;
typedef __attribute__((ext_vector_type(8)))  _Float16 v8h;
typedef __attribute__((ext_vector_type(8)))  float    v8f;

union ABfrag { v16h v; v8h h[2]; };

static __device__ __forceinline__ v8f wmma_f16(const ABfrag& a, const ABfrag& b, v8f c) {
  // v_wmma_f32_16x16x32_f16: D(16x16,f32) = A(16x32,f16) x B(32x16,f16) + C
  return __builtin_amdgcn_wmma_f32_16x16x32_f16(false, a.v, false, b.v,
                                                (short)0, c, false, false);
}

// ---------------------------------------------------------------------------
// fp32 -> fp16 weight conversion (run once per weight matrix)
// ---------------------------------------------------------------------------
__global__ void k_cvt_f16(const float* __restrict__ src, _Float16* __restrict__ dst, int n) {
  int i = blockIdx.x * 256 + threadIdx.x;
  if (i < n) dst[i] = (_Float16)src[i];
}

// ---------------------------------------------------------------------------
// xp0[t][i] = x[t]@W_ih0[i] + b_ih0[i] + b_hh0[i]     (K=6, pure VALU, mem-bound)
// ---------------------------------------------------------------------------
__global__ void k_xp0(const float* __restrict__ x, const float* __restrict__ Wih0,
                      const float* __restrict__ bih, const float* __restrict__ bhh,
                      float* __restrict__ xp) {
  int idx = blockIdx.x * 256 + threadIdx.x;      // T*H threads
  int t = idx >> 8, i = idx & (HH - 1);
  const float* xr = x + (size_t)t * II;
  const float* wr = Wih0 + (size_t)i * II;
  float s = bih[i] + bhh[i];
#pragma unroll
  for (int j = 0; j < II; ++j) s += xr[j] * wr[j];
  xp[idx] = s;
}

// ---------------------------------------------------------------------------
// Sequential ReLU recurrence, one workgroup (16 wave32), WMMA matvec per step.
//   h_t = relu(xp_t + W_hh h_{t-1})          (biases pre-folded into xp)
// Each wave owns ONE 16-column N-tile; B = W_hh^T fragments stay resident in
// VGPRs (8 K-tiles x 8 VGPRs = 64 VGPRs, no spills). Per step:
//   - xp is software-pipelined ONE FULL STEP ahead (load t+1 during step t,
//     carried across the barrier) so its global latency never serializes,
//   - A-fragment LDS loads rotate through 4 buffers so ds_load latency
//     overlaps the WMMA chain with partial DScnt waits,
//   - two accumulator chains (c0/c1) keep the XDL pipe fed,
//   - ping-pong h buffer => ONE barrier per step, and that barrier waits on
//     DScnt ONLY (inline asm): the hout global store is private streaming
//     output and must not stall the serial chain (s_endpgm drains it).
// ---------------------------------------------------------------------------
__global__ void __launch_bounds__(512, 1)
k_rnn(const float* __restrict__ xp,      // [T][H] f32, biases folded
      const _Float16* __restrict__ Whh,  // [H][H] f16 (row n = output unit n)
      _Float16* __restrict__ hout,       // [T][H] f16
      int T) {
  __shared__ alignas(16) _Float16 sh_h[2][HH];  // ping-pong hidden state (f16)

  const int tid  = threadIdx.x;              // 0..511
  const int lane = tid & 31;
  const int wv   = tid >> 5;                 // 0..15 -> owns N-tile wv
  const int nl   = lane & 15;                // N within tile / M within tile
  const int g    = lane >> 4;                // lane-half: selects K sub-range

  // ---- preload B = W_hh^T tiles into registers (f16) --------------------
  // B(32x16) layout: lane n%16 holds column n, K = 8*g + [0..7] and 16 + 8*g + [0..7] packed 2/VGPR.
  // B[k][n] = Whh[n][k] -> two 8-half runs of row n of Whh.
  ABfrag bt[8];
  const int ncol = wv * 16 + nl;             // global output column of this lane
#pragma unroll
  for (int kt = 0; kt < 8; ++kt) {
    const int kb = kt * 32 + 8 * g;
    bt[kt].h[0] = *(const v8h*)&Whh[(size_t)ncol * HH + kb];
    bt[kt].h[1] = *(const v8h*)&Whh[(size_t)ncol * HH + kb + 16];
  }

  // h_0 = 0
  if (tid < HH) { sh_h[0][tid] = (_Float16)0.0f; sh_h[1][tid] = (_Float16)0.0f; }
  __syncthreads();

  // xp pipelined one step ahead; lanes 16..31 duplicate lanes 0..15's
  // address (harmless) to avoid per-step EXEC save/restore around the load.
  float xpv = xp[ncol];                      // value for t = 0
  int p = 0;
  for (int t = 0; t < T; ++t) {
    const _Float16* hsrc = &sh_h[p][0];
    // Next step's xp: issued here, pinned above the barrier by its "memory"
    // clobber, consumed next iteration => full step of latency hiding.
    // (At t=T-1 this overreads into the adjacent ws region: mapped, harmless.)
    float xpn = xp[(size_t)(t + 1) * HH + ncol];
    __builtin_prefetch(&xp[(size_t)(t + 24) * HH + ncol], 0, 3);

    // A(16x32) with h broadcast across all 16 rows: lane needs
    // K = kt*32 + 8*g + [0..7] (VGPR0-3) and +16 (VGPR4-7): two aligned
    // ds_load_b128 per K-tile. Rotate through 4 buffers (issue 4 tiles ahead).
#define LDA(frag, kt)                                          \
    do {                                                       \
      const int _k0 = (kt) * 32 + 8 * g;                       \
      (frag).h[0] = *(const v8h*)&hsrc[_k0];                   \
      (frag).h[1] = *(const v8h*)&hsrc[_k0 + 16];              \
    } while (0)

    ABfrag a[4];
    v8f c0 = {0.f, 0.f, 0.f, 0.f, 0.f, 0.f, 0.f, 0.f};
    v8f c1 = {0.f, 0.f, 0.f, 0.f, 0.f, 0.f, 0.f, 0.f};
    LDA(a[0], 0);
    LDA(a[1], 1);
    LDA(a[2], 2);
    LDA(a[3], 3);
    c0 = wmma_f16(a[0], bt[0], c0); LDA(a[0], 4);
    c1 = wmma_f16(a[1], bt[1], c1); LDA(a[1], 5);
    c0 = wmma_f16(a[2], bt[2], c0); LDA(a[2], 6);
    c1 = wmma_f16(a[3], bt[3], c1); LDA(a[3], 7);
    c0 = wmma_f16(a[0], bt[4], c0);
    c1 = wmma_f16(a[1], bt[5], c1);
    c0 = wmma_f16(a[2], bt[6], c0);
    c1 = wmma_f16(a[3], bt[7], c1);
#undef LDA

    // D row M=0 lives in VGPR0 of lanes 0..15 (N = lane). Those lanes finish
    // the step for their column and write the *other* h buffer directly.
    if (lane < 16) {
      float v = c0[0] + c1[0] + xpv;
      v = v > 0.f ? v : 0.f;
      _Float16 hf = (_Float16)v;
      sh_h[p ^ 1][ncol] = hf;
      hout[(size_t)t * HH + ncol] = hf;
    }
    // Barrier that waits only on LDS ops (DScnt): the new h must be visible,
    // but the hout global store drains in the background.
    asm volatile("s_wait_dscnt 0x0\n\t"
                 "s_barrier_signal -1\n\t"
                 "s_barrier_wait -1" ::: "memory");
    xpv = xpn;
    p ^= 1;
  }
}

// ---------------------------------------------------------------------------
// xp1 = h1 @ W_ih1^T + (b_ih1 + b_hh1) : [T,H]x[H,H] f16 WMMA GEMM, f32 accum.
// One 16x16 output tile per wave; K=256 as 8 chunks of 32.
// ---------------------------------------------------------------------------
__global__ void k_gemm_proj(const _Float16* __restrict__ A,   // h1 [T][H] f16
                            const _Float16* __restrict__ Bw,  // Wih1 [H][H] f16
                            const float* __restrict__ bih,
                            const float* __restrict__ bhh,
                            float* __restrict__ xp) {
  const int lane = threadIdx.x & 31;
  const int wv   = threadIdx.x >> 5;                 // 4 waves / block
  const int tile = blockIdx.x * 4 + wv;              // (T/16)*(H/16) tiles
  const int tM = tile >> 4;                          // 0..T/16-1
  const int tN = tile & 15;                          // 0..15
  const int m  = lane & 15;
  const int g  = lane >> 4;
  const int row  = tM * 16 + m;                      // A row (time)
  const int ncol = tN * 16 + m;                      // B column (feature)

  v8f c = {0.f, 0.f, 0.f, 0.f, 0.f, 0.f, 0.f, 0.f};
#pragma unroll
  for (int kt = 0; kt < 8; ++kt) {
    const int k0 = kt * 32;
    ABfrag a, b;
    a.h[0] = *(const v8h*)&A[(size_t)row * HH + k0 + 8 * g];
    a.h[1] = *(const v8h*)&A[(size_t)row * HH + k0 + 16 + 8 * g];
    b.h[0] = *(const v8h*)&Bw[(size_t)ncol * HH + k0 + 8 * g];
    b.h[1] = *(const v8h*)&Bw[(size_t)ncol * HH + k0 + 16 + 8 * g];
    c = wmma_f16(a, b, c);
  }
  const float bias = bih[ncol] + bhh[ncol];
#pragma unroll
  for (int r = 0; r < 8; ++r) {
    const int rr = tM * 16 + r + 8 * g;              // C layout: M = r + 8*half
    xp[(size_t)rr * HH + ncol] = c[r] + bias;
  }
}

// ---------------------------------------------------------------------------
// logits[t][c] = h2[t] @ W_out[c] + b_out[c]   (K=256, C=10; W_out in LDS)
// ---------------------------------------------------------------------------
__global__ void k_logits(const _Float16* __restrict__ h2, const float* __restrict__ Wout,
                         const float* __restrict__ bout, float* __restrict__ logits) {
  __shared__ float sw[CC * HH];
  __shared__ float sb[CC];
  for (int i = threadIdx.x; i < CC * HH; i += 256) sw[i] = Wout[i];
  if (threadIdx.x < CC) sb[threadIdx.x] = bout[threadIdx.x];
  __syncthreads();

  const int t = blockIdx.x * 256 + threadIdx.x;
  const _Float16* hr = h2 + (size_t)t * HH;
  float acc[CC];
#pragma unroll
  for (int c = 0; c < CC; ++c) acc[c] = sb[c];
  for (int j = 0; j < HH; j += 8) {
    v8h hv8 = *(const v8h*)&hr[j];
#pragma unroll
    for (int u = 0; u < 8; ++u) {
      float hv = (float)hv8[u];
#pragma unroll
      for (int c = 0; c < CC; ++c) acc[c] += hv * sw[c * HH + j + u];
    }
  }
#pragma unroll
  for (int c = 0; c < CC; ++c) logits[(size_t)t * CC + c] = acc[c];
}

// ---------------------------------------------------------------------------
// Per-class (time-axis) softmax statistics: stats[c]=max, stats[C+c]=sum exp
// ---------------------------------------------------------------------------
__global__ void k_colstats(const float* __restrict__ logits, float* __restrict__ stats, int T) {
  const int c = blockIdx.x;
  __shared__ float red[256];
  float mx = -1e30f;
  for (int t = threadIdx.x; t < T; t += 256) mx = fmaxf(mx, logits[(size_t)t * CC + c]);
  red[threadIdx.x] = mx;
  __syncthreads();
  for (int s = 128; s > 0; s >>= 1) {
    if (threadIdx.x < s) red[threadIdx.x] = fmaxf(red[threadIdx.x], red[threadIdx.x + s]);
    __syncthreads();
  }
  mx = red[0];
  __syncthreads();
  float sum = 0.f;
  for (int t = threadIdx.x; t < T; t += 256) sum += __expf(logits[(size_t)t * CC + c] - mx);
  red[threadIdx.x] = sum;
  __syncthreads();
  for (int s = 128; s > 0; s >>= 1) {
    if (threadIdx.x < s) red[threadIdx.x] += red[threadIdx.x + s];
    __syncthreads();
  }
  if (threadIdx.x == 0) { stats[c] = mx; stats[CC + c] = red[0]; }
}

__global__ void k_softmax_write(const float* __restrict__ logits, const float* __restrict__ stats,
                                float* __restrict__ out) {
  int idx = blockIdx.x * 256 + threadIdx.x;   // T*C
  int c = idx % CC;
  out[idx] = __expf(logits[idx] - stats[c]) / stats[CC + c];
}

// ---------------------------------------------------------------------------
// Host launch
// ---------------------------------------------------------------------------
extern "C" void kernel_launch(void* const* d_in, const int* in_sizes, int n_in,
                              void* d_out, int out_size, void* d_ws, size_t ws_size,
                              hipStream_t stream) {
  (void)in_sizes; (void)n_in; (void)out_size; (void)ws_size;

  const float* x     = (const float*)d_in[0];
  const float* Wih0  = (const float*)d_in[1];
  const float* Whh0  = (const float*)d_in[2];
  const float* bih0  = (const float*)d_in[3];
  const float* bhh0  = (const float*)d_in[4];
  const float* Wih1  = (const float*)d_in[5];
  const float* Whh1  = (const float*)d_in[6];
  const float* bih1  = (const float*)d_in[7];
  const float* bhh1  = (const float*)d_in[8];
  const float* Wout  = (const float*)d_in[9];
  const float* bout  = (const float*)d_in[10];
  float* out = (float*)d_out;

  char* ws = (char*)d_ws;
  const size_t MB = 1024ull * 1024ull;
  float*     xp       = (float*)(ws);                        // T*H f32 = 64MB (reused for xp0/xp1)
  _Float16*  h1       = (_Float16*)(ws + 64 * MB);           // T*H f16 = 32MB
  _Float16*  h2       = (_Float16*)(ws + 96 * MB);           // T*H f16 = 32MB
  float*     logits   = (float*)(ws + 128 * MB);             // T*C f32 ~ 2.6MB
  float*     stats    = (float*)(ws + 131 * MB);             // 2*C f32
  _Float16*  Whh0_16  = (_Float16*)(ws + 132 * MB);          // 128KB each
  _Float16*  Wih1_16  = (_Float16*)(ws + 132 * MB + 256 * 1024);
  _Float16*  Whh1_16  = (_Float16*)(ws + 132 * MB + 512 * 1024);

  const int HW = HH * HH;  // 65536

  // 1) weight conversions to f16
  k_cvt_f16<<<HW / 256, 256, 0, stream>>>(Whh0, Whh0_16, HW);
  k_cvt_f16<<<HW / 256, 256, 0, stream>>>(Wih1, Wih1_16, HW);
  k_cvt_f16<<<HW / 256, 256, 0, stream>>>(Whh1, Whh1_16, HW);

  // 2) layer-0 input projection (biases folded)
  k_xp0<<<(TT * HH) / 256, 256, 0, stream>>>(x, Wih0, bih0, bhh0, xp);

  // 3) layer-0 recurrence (single WGP, WMMA matvec, W_hh resident in VGPRs)
  k_rnn<<<1, 512, 0, stream>>>(xp, Whh0_16, h1, TT);

  // 4) layer-1 input projection: WMMA GEMM [T,H]x[H,H], biases folded
  k_gemm_proj<<<(TT / 16) * (HH / 16) / 4, 128, 0, stream>>>(h1, Wih1_16, bih1, bhh1, xp);

  // 5) layer-1 recurrence
  k_rnn<<<1, 512, 0, stream>>>(xp, Whh1_16, h2, TT);

  // 6) output head + time-axis softmax
  k_logits<<<TT / 256, 256, 0, stream>>>(h2, Wout, bout, logits);
  k_colstats<<<CC, 256, 0, stream>>>(logits, stats, TT);
  k_softmax_write<<<(TT * CC) / 256, 256, 0, stream>>>(logits, stats, out);
}
